// StandAloneSelfAttention_26156350832781
// MI455X (gfx1250) — compile-verified, matrix operand order checked by reading, not run on hardware
//
#include <hip/hip_runtime.h>
#include <hip/hip_bf16.h>

// ---------------- problem constants (from reference) ----------------
#define BB    2
#define CIN   128
#define HH    56
#define WW    56
#define COUT  256
#define HEADS 8
#define DH    32
#define KW    7

#define HP    68                 // 56 + 2*6 (pad by 6: covers q conv and pre-padded k/v conv)
#define HKV   62                 // k/v conv output extent
#define KDIM  (KW * KW * CIN)    // 6272 GEMM K, ordered (ky,kx,c)
#define NPIX_Q  (BB * HH * WW)   // 6272
#define NPIX_KV (BB * HKV * HKV) // 7688

typedef __attribute__((ext_vector_type(16))) __bf16 v16bf;
typedef __attribute__((ext_vector_type(8)))  __bf16 v8bf;
typedef __attribute__((ext_vector_type(8)))  float  v8f;

__device__ __forceinline__ __bf16 f32_to_bf16(float f) {
    unsigned u = __builtin_bit_cast(unsigned, f);
    u += 0x7FFFu + ((u >> 16) & 1u);            // round-to-nearest-even
    unsigned short s = (unsigned short)(u >> 16);
    return __builtin_bit_cast(__bf16, s);
}

// --------- CDNA5 async global->LDS copy (ASYNCcnt-tracked, §15.18.3 op 98) ---------
typedef __attribute__((address_space(3))) void lds_void;

__device__ __forceinline__ unsigned lds_offset(const void* p) {
    // generic -> LDS addrspace -> 32-bit LDS byte offset
    return (unsigned)(unsigned long long)(lds_void*)p;
}

__device__ __forceinline__ void async_copy_b128(unsigned dst_lds, const void* src) {
    asm volatile("global_load_async_to_lds_b128 %0, %1, off"
                 :: "v"(dst_lds), "v"((unsigned long long)src)
                 : "memory");
}

__device__ __forceinline__ void wait_asynccnt0() {
    asm volatile("s_wait_asynccnt 0x0" ::: "memory");
}

// ---------------- stage 1: pad input by 6, convert to bf16 NHWC ----------------
// xpad[((b*HP + yy)*HP + xx)*CIN + c]
__global__ __launch_bounds__(256) void pad_input_bf16(const float* __restrict__ in,
                                                      __bf16* __restrict__ xpad) {
    long t = (long)blockIdx.x * blockDim.x + threadIdx.x;  // over B*HP*HP*CIN
    const long total = (long)BB * HP * HP * CIN;
    if (t >= total) return;
    int c  = (int)(t & (CIN - 1));
    long r = t >> 7;                 // CIN = 128
    int xx = (int)(r % HP);
    r /= HP;
    int yy = (int)(r % HP);
    int b0 = (int)(r / HP);
    float v = 0.f;
    int yi = yy - 6, xi = xx - 6;
    if (yi >= 0 && yi < HH && xi >= 0 && xi < WW)
        v = in[(((long)b0 * CIN + c) * HH + yi) * WW + xi];
    xpad[t] = f32_to_bf16(v);
}

// ---------------- stage 2: weights f32 [Cout,Cin,7,7] -> bf16 [Cout][tap][c] ----------------
__global__ __launch_bounds__(256) void convert_w_bf16(const float* __restrict__ w,
                                                      __bf16* __restrict__ wb) {
    long t = (long)blockIdx.x * blockDim.x + threadIdx.x;  // over COUT*KDIM (output index)
    const long total = (long)COUT * KDIM;
    if (t >= total) return;
    int m   = (int)(t / KDIM);
    int r   = (int)(t - (long)m * KDIM);
    int tap = r >> 7;                // (ky*7+kx)
    int c   = r & (CIN - 1);
    wb[t] = f32_to_bf16(w[((long)m * CIN + c) * (KW * KW) + tap]);
}

// ---------------- stage 3: implicit-GEMM conv via WMMA + async-LDS fill ----------------
// Block = 256 threads = 8 waves; block tile = 64 Cout x 128 pixels.
// Wave tile = 32 Cout x 32 pixels -> 4 WMMAs per K-step with 2x2 fragment reuse.
// K ordered (tap, c): a 32-wide K-step sits inside one 7x7 tap, so all fills are
// contiguous 16-byte GLOBAL_LOAD_ASYNC_TO_LDS_B128 copies from the NHWC padded
// input / packed weights. LDS double-buffered: while buffer `buf` feeds the
// WMMAs, async copies fill `buf^1`; s_wait_asynccnt + barrier ends the step.
__global__ __launch_bounds__(256) void conv_wmma(const __bf16* __restrict__ xpad,
                                                 const __bf16* __restrict__ wb,
                                                 float* __restrict__ out,
                                                 int H0, int W0, int base, int Npix) {
    __shared__ alignas(32) __bf16 Alds[2][64][48];    // 64 cout x 32 K (+pad) : 12 KB
    __shared__ alignas(32) __bf16 Blds[2][128][48];   // 128 pix x 32 K (+pad) : 24 KB

    const int tid   = threadIdx.x;
    const int lane  = tid & 31;
    const int wave  = tid >> 5;
    const int wr    = wave >> 2;        // cout half 0..1 (32 couts each)
    const int wp    = wave & 3;         // pixel quarter 0..3 (32 pixels each)
    const int tileN = blockIdx.x;       // 128-pixel tile
    const int tileM = blockIdx.y;       // 64-cout tile

    v8f acc00 = {}, acc01 = {}, acc10 = {}, acc11 = {};

    // A cooperative fill: 64x32 bf16 = 2048 elems -> 1 async b128 per thread
    const int arow = tid >> 2;          // 0..63
    const int acol = (tid & 3) * 8;     // 0,8,16,24
    const long awbase = (long)(tileM * 64 + arow) * KDIM + acol;

    // B cooperative fill: 128x32 bf16 = 4096 elems -> 2 async b128 per thread
    const int prow = tid >> 2;          // 0..63 (rows prow and prow+64)
    const int pcol = (tid & 3) * 8;
    int pb[2], pyy[2], pxx[2];
#pragma unroll
    for (int j = 0; j < 2; ++j) {
        int n = tileN * 128 + prow + j * 64;
        if (n >= Npix) n = Npix - 1;    // clamp (stores are guarded)
        int b0 = n / (H0 * W0);
        int r  = n - b0 * H0 * W0;
        int yy = r / W0;
        pb[j]  = b0;
        pyy[j] = yy + base;
        pxx[j] = (r - yy * W0) + base;
    }

    // async fill of K-step `kn` into LDS buffer `nb`
    auto issue_fill = [&](int kn, int nb) {
        int tap = kn >> 7;              // (ky*7+kx), CIN = 128
        int c0  = kn & (CIN - 1);
        int ky  = tap / KW;
        int kx  = tap - ky * KW;
        async_copy_b128(lds_offset(&Alds[nb][arow][acol]), wb + awbase + kn);
#pragma unroll
        for (int j = 0; j < 2; ++j) {
            const __bf16* src = xpad +
                (((long)pb[j] * HP + (pyy[j] + ky)) * HP + (pxx[j] + kx)) * CIN + c0 + pcol;
            async_copy_b128(lds_offset(&Blds[nb][prow + j * 64][pcol]), src);
        }
    };

    // prologue: fill buffer 0 with K-step 0
    issue_fill(0, 0);
    wait_asynccnt0();
    __syncthreads();

    int buf = 0;
    for (int kk0 = 0; kk0 < KDIM; kk0 += 32) {
        int kn = kk0 + 32;
        if (kn < KDIM) issue_fill(kn, buf ^ 1);   // overlaps with the WMMAs below

        v16bf a0 = *(const v16bf*)(&Alds[buf][wr * 32 + (lane & 15)][(lane >> 4) * 16]);
        v16bf a1 = *(const v16bf*)(&Alds[buf][wr * 32 + 16 + (lane & 15)][(lane >> 4) * 16]);
        v16bf b0 = *(const v16bf*)(&Blds[buf][wp * 32 + (lane & 15)][(lane >> 4) * 16]);
        v16bf b1 = *(const v16bf*)(&Blds[buf][wp * 32 + 16 + (lane & 15)][(lane >> 4) * 16]);
        acc00 = __builtin_amdgcn_wmma_f32_16x16x32_bf16(false, a0, false, b0, (short)0, acc00, false, false);
        acc01 = __builtin_amdgcn_wmma_f32_16x16x32_bf16(false, a0, false, b1, (short)0, acc01, false, false);
        acc10 = __builtin_amdgcn_wmma_f32_16x16x32_bf16(false, a1, false, b0, (short)0, acc10, false, false);
        acc11 = __builtin_amdgcn_wmma_f32_16x16x32_bf16(false, a1, false, b1, (short)0, acc11, false, false);

        wait_asynccnt0();               // next buffer's fill landed
        __syncthreads();
        buf ^= 1;
    }

    // store C fragments (layout: lanes 0-15 -> M=i, lanes 16-31 -> M=i+8; N = lane&15)
    const long mstride = (long)H0 * W0;
#pragma unroll
    for (int nt = 0; nt < 2; ++nt) {
        int ncol = tileN * 128 + wp * 32 + nt * 16 + (lane & 15);
        if (ncol < Npix) {
            int b0 = ncol / (H0 * W0);
            int r  = ncol - b0 * H0 * W0;
            int y  = r / W0;
            int x  = r - y * W0;
            int m0 = tileM * 64 + wr * 32 + ((lane >> 4) << 3);
            long obase = (((long)b0 * COUT + m0) * H0 + y) * W0 + x;
            v8f accA = nt ? acc01 : acc00;
            v8f accB = nt ? acc11 : acc10;
#pragma unroll
            for (int i = 0; i < 8; ++i)
                out[obase + i * mstride] = accA[i];
            obase += 16 * mstride;
#pragma unroll
            for (int i = 0; i < 8; ++i)
                out[obase + i * mstride] = accB[i];
        }
    }
}

// ---------------- stage 4: windowed attention (online softmax) ----------------
__global__ __launch_bounds__(256) void attn_window(const float* __restrict__ qf,
                                                   const float* __restrict__ kf,
                                                   const float* __restrict__ vf,
                                                   const float* __restrict__ rel_h,
                                                   const float* __restrict__ rel_w,
                                                   float* __restrict__ out) {
    long t = (long)blockIdx.x * blockDim.x + threadIdx.x;   // over B*COUT*H*W
    const long total = (long)BB * COUT * HH * WW;
    if (t >= total) return;
    const int HWp = HH * WW;
    int b0 = (int)(t / ((long)COUT * HWp));
    int r  = (int)(t - (long)b0 * COUT * HWp);
    int c  = r / HWp;
    int p  = r - c * HWp;
    int y  = p / WW;
    int x  = p - y * WW;

    const float q = qf[t];                                  // same [B,COUT,H,W] layout
    const long kvbase = ((long)b0 * COUT + c) * (HKV * HKV);

    float m = -1e30f, s = 0.f, o = 0.f;
#pragma unroll
    for (int ky = 0; ky < KW; ++ky) {
#pragma unroll
        for (int kx = 0; kx < KW; ++kx) {
            long idx = kvbase + (long)(y + ky) * HKV + (x + kx);
            float kval = kf[idx];
            kval += (c < COUT / 2) ? rel_h[c * KW + ky]
                                   : rel_w[(c - COUT / 2) * KW + kx];
            float logit = q * kval;
            float vval  = vf[idx];
            if (logit > m) {
                float sc = __expf(m - logit);
                s = s * sc + 1.f;
                o = o * sc + vval;
                m = logit;
            } else {
                float e = __expf(logit - m);
                s += e;
                o += e * vval;
            }
        }
    }
    // out channel permutation: [B,heads,dh,..] -> transpose(0,2,1,..) -> [B,COUT,..]
    int cOut = (c % DH) * HEADS + (c / DH);
    out[(((long)b0 * COUT + cOut) * HH + y) * WW + x] = o / s;
}

// ---------------- launcher ----------------
extern "C" void kernel_launch(void* const* d_in, const int* in_sizes, int n_in,
                              void* d_out, int out_size, void* d_ws, size_t ws_size,
                              hipStream_t stream) {
    const float* x   = (const float*)d_in[0];
    const float* Wq  = (const float*)d_in[1];
    const float* Wk  = (const float*)d_in[2];
    const float* Wv  = (const float*)d_in[3];
    const float* rh  = (const float*)d_in[4];
    const float* rw  = (const float*)d_in[5];
    float* out       = (float*)d_out;

    // workspace carve-out (deterministic, pure arithmetic)
    char*  ws  = (char*)d_ws;
    size_t off = 0;
    auto take = [&](size_t bytes) -> char* {
        char* p = ws + off;
        off = (off + bytes + 255) & ~(size_t)255;
        return p;
    };
    const size_t xpadElems = (size_t)BB * HP * HP * CIN;
    const size_t wElems    = (size_t)COUT * KDIM;
    __bf16* xpad = (__bf16*)take(xpadElems * 2);
    __bf16* wqb  = (__bf16*)take(wElems * 2);
    __bf16* wkb  = (__bf16*)take(wElems * 2);
    __bf16* wvb  = (__bf16*)take(wElems * 2);
    float*  qf   = (float*)take((size_t)BB * COUT * HH * WW * 4);
    float*  kfm  = (float*)take((size_t)BB * COUT * HKV * HKV * 4);
    float*  vfm  = (float*)take((size_t)BB * COUT * HKV * HKV * 4);

    // stage 1: pad + convert input to NHWC bf16
    {
        long nTot = (long)xpadElems;
        pad_input_bf16<<<(unsigned)((nTot + 255) / 256), 256, 0, stream>>>(x, xpad);
    }
    // stage 2: convert + permute weights
    {
        long nTot = (long)wElems;
        unsigned g = (unsigned)((nTot + 255) / 256);
        convert_w_bf16<<<g, 256, 0, stream>>>(Wq, wqb);
        convert_w_bf16<<<g, 256, 0, stream>>>(Wk, wkb);
        convert_w_bf16<<<g, 256, 0, stream>>>(Wv, wvb);
    }
    // stage 3: the three convs as implicit GEMM + WMMA
    {
        dim3 gq((NPIX_Q + 127) / 128, COUT / 64);
        conv_wmma<<<gq, 256, 0, stream>>>(xpad, wqb, qf, HH, WW, /*base=*/3, NPIX_Q);
        dim3 gkv((NPIX_KV + 127) / 128, COUT / 64);
        conv_wmma<<<gkv, 256, 0, stream>>>(xpad, wkb, kfm, HKV, HKV, /*base=*/0, NPIX_KV);
        conv_wmma<<<gkv, 256, 0, stream>>>(xpad, wvb, vfm, HKV, HKV, /*base=*/0, NPIX_KV);
    }
    // stage 4: attention epilogue
    {
        long nTot = (long)BB * COUT * HH * WW;
        attn_window<<<(unsigned)((nTot + 255) / 256), 256, 0, stream>>>(qf, kfm, vfm, rh, rw, out);
    }
}
